// Transformer_31327491457655
// MI455X (gfx1250) — compile-verified, matrix operand order checked by reading, not run on hardware
//
#include <hip/hip_runtime.h>
#include <math.h>

// Model constants (fixed by the reference)
#define BB 2
#define SS 2048
#define DD 512
#define HH 8
#define DKK 64
#define DVV 64
#define DFF 2048
#define NLL 4
#define VOC 8000

typedef __attribute__((ext_vector_type(16))) _Float16 v16h;
typedef __attribute__((ext_vector_type(8)))  _Float16 half8;
typedef __attribute__((ext_vector_type(8)))  float    v8f;
typedef __attribute__((ext_vector_type(4)))  unsigned v4u;
typedef __attribute__((ext_vector_type(8)))  int      v8i;
typedef __attribute__((ext_vector_type(4)))  int      v4i;

// ---------------------------------------------------------------------------
// TDM (Tensor Data Mover) support: one wave-level instruction DMAs a 2D tile
// (f16, contiguous innermost dim, arbitrary row stride) into LDS; tracked with
// TENSORcnt. This toolchain exposes the 6-arg builtin form:
//   (uint32x4 g0, int32x8 g1, int32x4 g2, int32x4 g3, int32x8, i32 cpol)
// ---------------------------------------------------------------------------
#if defined(__has_builtin)
#  if __has_builtin(__builtin_amdgcn_tensor_load_to_lds) && \
      __has_builtin(__builtin_amdgcn_s_wait_tensorcnt)
#    define HAVE_TDM 1
#  endif
#endif
#ifndef HAVE_TDM
#  define HAVE_TDM 0
#endif

__device__ __forceinline__ unsigned lds_off_of(const void* p) {
  // LDS aperture lives in addr[63:32]; addr[31:0] is the LDS byte offset.
  return (unsigned)(unsigned long long)(size_t)p;
}

#if HAVE_TDM
// D# per CDNA5 ISA: group0 = {count/flags, lds_addr,
// global_addr, type=2}; group1 = {data_size, tensor dims, tile dims, stride}.
// tensor_dim0/1 set huge (1<<24) so no OOB zero-fill kicks in.
__device__ __forceinline__ void tdm_load_2d_f16(unsigned lds_off,
                                                const _Float16* g,
                                                unsigned tile_cols,
                                                unsigned tile_rows,
                                                unsigned stride_elems) {
  unsigned long long ga = (unsigned long long)(size_t)g;
  v4u g0;
  g0.x = 1u;                                               // count=1, user mode
  g0.y = lds_off;                                          // LDS byte address
  g0.z = (unsigned)ga;                                     // global addr [31:0]
  g0.w = (unsigned)((ga >> 32) & 0x1FFFFFFull) | (2u << 30); // [56:32] | type=2
  v8i g1;
  g1[0] = (int)(1u << 16);                // wg_mask=0, data_size=1 (2 bytes)
  g1[1] = 0;                              // tensor_dim0 = 1<<24 -> low16 = 0
  g1[2] = 0x100;                          // dim0 hi16=0x100; dim1 low16=0
  g1[3] = (int)(0x100u | (tile_cols << 16)); // dim1 hi16=0x100; tile_dim0
  g1[4] = (int)tile_rows;                 // tile_dim1; tile_dim2=0 (2D)
  g1[5] = (int)stride_elems;              // tensor_dim0_stride [31:0]
  g1[6] = 0;
  g1[7] = 0;
  v4i z4 = {0, 0, 0, 0};
  v8i z8 = {0, 0, 0, 0, 0, 0, 0, 0};
  __builtin_amdgcn_tensor_load_to_lds(g0, g1, z4, z4, z8, 0);
}
#endif

// Cooperative fallback 2D tile copy (f16, b128 moves).
__device__ __forceinline__ void tile_copy(_Float16* dst, const _Float16* src,
                                          int rows, int cols, int sstride,
                                          int tid, int nthr) {
  int slots = rows * cols / 8;
  int cw = cols / 8;
  for (int i = tid; i < slots; i += nthr) {
    int r = i / cw;
    int c = (i - r * cw) * 8;
    *(half8*)(dst + r * cols + c) = *(const half8*)(src + (size_t)r * sstride + c);
  }
}

// ---------------------------------------------------------------------------
// WMMA helpers
// ---------------------------------------------------------------------------
__device__ __forceinline__ v8f wmma_f16(v16h a, v16h b, v8f c) {
  return __builtin_amdgcn_wmma_f32_16x16x32_f16(
      false, a, false, b, (short)0, c, false, false);
}

// 16x32 f16 fragment load (A-layout; B uses the symmetric layout from a
// K-major LDS tile). Per lane: halves 0..7 hold K=8*(lane>>4)+e, halves 8..15
// hold K=16+8*(lane>>4)+e -> two contiguous ds_load_b128.
__device__ __forceinline__ v16h lds_frag(const _Float16* base, int row,
                                         int stride, int kbase, int hi8) {
  union { v16h v; half8 h[2]; } u;
  const _Float16* p = base + row * stride + kbase + hi8;
  u.h[0] = *(const half8*)(p);
  u.h[1] = *(const half8*)(p + 16);
  return u.v;
}

// ---------------------------------------------------------------------------
// Embedding + sinusoidal positional encoding -> fp32 x and f16 mirror
// ---------------------------------------------------------------------------
__global__ __launch_bounds__(256) void embed_pe_kernel(
    const int* __restrict__ tok, const float* __restrict__ emb,
    float* __restrict__ out, _Float16* __restrict__ outh) {
  int idx = blockIdx.x * 256 + threadIdx.x;   // over BB*SS*DD
  int row = idx >> 9;                         // DD = 512
  int d   = idx & (DD - 1);
  int s   = row & (SS - 1);
  int t   = tok[row];
  float e   = (float)(2 * (d >> 1)) / (float)DD;
  float ang = (float)s * __powf(10000.0f, -e);
  float pe  = (d & 1) ? __cosf(ang) : __sinf(ang);
  float v   = emb[(size_t)t * DD + d] + pe;
  out[idx]  = v;
  outh[idx] = (_Float16)v;
}

// ---------------------------------------------------------------------------
// Weight convert+transpose: W[K,N] fp32 -> Wt[N,K] f16 (once per call)
// ---------------------------------------------------------------------------
__global__ __launch_bounds__(256) void wt_kernel(const float* __restrict__ w,
                                                 _Float16* __restrict__ wt,
                                                 int K, int N) {
  int idx = blockIdx.x * 256 + threadIdx.x;   // over N*K
  int n = idx / K;
  int k = idx - n * K;
  wt[idx] = (_Float16)w[(size_t)k * N + n];
}

// ---------------------------------------------------------------------------
// GEMM: out[M,N] = A[M,K](f16) @ Wt[N,K](f16)^T (+bias) (opt relu)
// 64x64 tile / 8 waves / K-step 32. TDM double-buffered tile loads.
// outmode: 0 = fp32 row-major, 1 = f16 row-major, 2 = f16 V-transposed
// ([b][h][dv][s] layout for the attention V operand).
// ---------------------------------------------------------------------------
__global__ __launch_bounds__(256) void gemm_kernel(
    const _Float16* __restrict__ A, const _Float16* __restrict__ Wt,
    const float* __restrict__ bias, void* __restrict__ out,
    int M, int K, int N, int relu, int outmode) {
  __shared__ _Float16 As[2][64 * 32];   // [m][k]
  __shared__ _Float16 Bs[2][64 * 32];   // [n][k]

  const int tid  = threadIdx.x;
  const int lane = tid & 31;
  const int wave = tid >> 5;
  const int wm   = wave >> 1;
  const int wn   = wave & 1;
  const int hi8  = (lane >> 4) << 3;
  const int ln   = lane & 15;
  const int Mb   = blockIdx.y * 64;
  const int Nb   = blockIdx.x * 64;
  const int nIter = K >> 5;

  v8f c0 = {}; v8f c1 = {};

#if HAVE_TDM
  if (wave == 0) {
    tdm_load_2d_f16(lds_off_of(As[0]), A + (size_t)Mb * K, 32, 64, K);
    tdm_load_2d_f16(lds_off_of(Bs[0]), Wt + (size_t)Nb * K, 32, 64, K);
  }
#else
  tile_copy(As[0], A + (size_t)Mb * K, 64, 32, K, tid, 256);
  tile_copy(Bs[0], Wt + (size_t)Nb * K, 64, 32, K, tid, 256);
#endif

  for (int i = 0; i < nIter; ++i) {
    const int p = i & 1;
#if HAVE_TDM
    if (wave == 0) {
      if (i + 1 < nIter) {
        const int k0n = (i + 1) << 5;
        tdm_load_2d_f16(lds_off_of(As[p ^ 1]), A + (size_t)Mb * K + k0n, 32, 64, K);
        tdm_load_2d_f16(lds_off_of(Bs[p ^ 1]), Wt + (size_t)Nb * K + k0n, 32, 64, K);
        __builtin_amdgcn_s_wait_tensorcnt(2);   // oldest 2 (current buf) done
      } else {
        __builtin_amdgcn_s_wait_tensorcnt(0);
      }
    }
#else
    if (i + 1 < nIter) {
      const int k0n = (i + 1) << 5;
      tile_copy(As[p ^ 1], A + (size_t)Mb * K + k0n, 64, 32, K, tid, 256);
      tile_copy(Bs[p ^ 1], Wt + (size_t)Nb * K + k0n, 64, 32, K, tid, 256);
    }
#endif
    __syncthreads();                 // buf p ready for all waves
    v16h a  = lds_frag(As[p], wm * 16 + ln,      32, 0, hi8);
    v16h b0 = lds_frag(Bs[p], wn * 32 + ln,      32, 0, hi8);
    v16h b1 = lds_frag(Bs[p], wn * 32 + 16 + ln, 32, 0, hi8);
    c0 = wmma_f16(a, b0, c0);
    c1 = wmma_f16(a, b1, c1);
    __syncthreads();                 // reads of buf p done before overwrite
  }

  // Epilogue. C layout: element r -> row = r + 8*(lane>>4), col = lane&15.
  const int rbase = Mb + wm * 16 + hi8;
  const int col0  = Nb + wn * 32 + ln;
  const int col1  = col0 + 16;
  const float bs0 = bias ? bias[col0] : 0.0f;
  const float bs1 = bias ? bias[col1] : 0.0f;
  for (int r = 0; r < 8; ++r) {
    float x0 = c0[r] + bs0;
    float x1 = c1[r] + bs1;
    if (relu) { x0 = fmaxf(x0, 0.0f); x1 = fmaxf(x1, 0.0f); }
    const int rg = rbase + r;
    if (outmode == 0) {
      float* o = (float*)out;
      o[(size_t)rg * N + col0] = x0;
      o[(size_t)rg * N + col1] = x1;
    } else if (outmode == 1) {
      _Float16* o = (_Float16*)out;
      o[(size_t)rg * N + col0] = (_Float16)x0;
      o[(size_t)rg * N + col1] = (_Float16)x1;
    } else {
      // V-transposed: vth[((b*H + h)*64 + dv)*S + s]
      _Float16* o = (_Float16*)out;
      const int b = rg >> 11, s = rg & (SS - 1);
      o[((size_t)(b * HH + (col0 >> 6)) * 64 + (col0 & 63)) * SS + s] = (_Float16)x0;
      o[((size_t)(b * HH + (col1 >> 6)) * 64 + (col1 & 63)) * SS + s] = (_Float16)x1;
    }
  }
}

// ---------------------------------------------------------------------------
// Flash-style attention (no 1/sqrt(dk) scale, matching the reference).
// q,k: [B,S,H*64] f16. vt: [B,H,64,S] f16 (pre-transposed by the V GEMM).
// out: [B,S,H*64] f16. One block = (b,h,64-row q tile), 4 waves.
// ---------------------------------------------------------------------------
__global__ __launch_bounds__(128) void attn_kernel(
    const _Float16* __restrict__ q, const _Float16* __restrict__ k,
    const _Float16* __restrict__ vt, _Float16* __restrict__ out, int causal) {
  __shared__ _Float16 Qs[64 * 64];       // [qrow][dk]
  __shared__ _Float16 Ks[64 * 64];       // [key][dk]
  __shared__ _Float16 Vt[64 * 64];       // [dv][key]
  __shared__ _Float16 Ps[4 * 16 * 64];   // per-wave P tile [qrow][key]

  const int tid  = threadIdx.x;
  const int lane = tid & 31;
  const int w    = tid >> 5;
  const int hi8  = (lane >> 4) << 3;
  const int ln   = lane & 15;
  const int qt = blockIdx.x, h = blockIdx.y, b = blockIdx.z;
  const int SD = HH * DKK;               // 512

  const _Float16* qg = q  + ((size_t)b * SS + qt * 64) * SD + h * DKK;
  const _Float16* vb = vt + ((size_t)(b * HH + h) * 64) * SS;   // + dv*S + key

#if HAVE_TDM
  if (tid < 32) tdm_load_2d_f16(lds_off_of(Qs), qg, 64, 64, SD);
#else
  tile_copy(Qs, qg, 64, 64, SD, tid, 128);
#endif

  float m_run[8], l_run[8];
  v8f o[4] = {{}, {}, {}, {}};
  for (int r = 0; r < 8; ++r) { m_run[r] = -1e30f; l_run[r] = 0.0f; }

  const int ktmax = causal ? (qt + 1) : (SS / 64);
  for (int kt = 0; kt < ktmax; ++kt) {
    __syncthreads();                     // previous reads of Ks/Vt done
    const _Float16* kg = k + ((size_t)b * SS + kt * 64) * SD + h * DKK;
#if HAVE_TDM
    if (tid < 32) {
      tdm_load_2d_f16(lds_off_of(Ks), kg, 64, 64, SD);
      tdm_load_2d_f16(lds_off_of(Vt), vb + kt * 64, 64, 64, SS);
      __builtin_amdgcn_s_wait_tensorcnt(0);   // also covers the Q load
    }
#else
    tile_copy(Ks, kg, 64, 64, SD, tid, 128);
    tile_copy(Vt, vb + kt * 64, 64, 64, SS, tid, 128);
#endif
    __syncthreads();

    // Scores: 16 rows x 64 keys per wave, contraction over dk=64.
    v8f sc[4] = {{}, {}, {}, {}};
    v16h a0 = lds_frag(Qs, w * 16 + ln, 64, 0,  hi8);
    v16h a1 = lds_frag(Qs, w * 16 + ln, 64, 32, hi8);
    for (int ns = 0; ns < 4; ++ns) {
      v16h bb0 = lds_frag(Ks, ns * 16 + ln, 64, 0,  hi8);
      v16h bb1 = lds_frag(Ks, ns * 16 + ln, 64, 32, hi8);
      sc[ns] = wmma_f16(a0, bb0, sc[ns]);
      sc[ns] = wmma_f16(a1, bb1, sc[ns]);
    }

    if (causal && kt == qt) {            // diagonal tile mask
      const int row0 = qt * 64 + w * 16 + hi8;
      for (int ns = 0; ns < 4; ++ns) {
        const int col = kt * 64 + ns * 16 + ln;
        for (int r = 0; r < 8; ++r)
          if (col > row0 + r) sc[ns][r] = -1e30f;
      }
    }

    // Row max over this tile's 64 keys
    float rmax[8];
    for (int r = 0; r < 8; ++r)
      rmax[r] = fmaxf(fmaxf(sc[0][r], sc[1][r]), fmaxf(sc[2][r], sc[3][r]));
    for (int m2 = 1; m2 <= 8; m2 <<= 1)
      for (int r = 0; r < 8; ++r)
        rmax[r] = fmaxf(rmax[r], __shfl_xor(rmax[r], m2, 32));

    // Online softmax update
    float pscale[8];
    for (int r = 0; r < 8; ++r) {
      float mn = fmaxf(m_run[r], rmax[r]);
      pscale[r] = __expf(m_run[r] - mn);
      m_run[r] = mn;
      l_run[r] *= pscale[r];
    }
    for (int ns = 0; ns < 4; ++ns)
      for (int r = 0; r < 8; ++r)
        sc[ns][r] = __expf(sc[ns][r] - m_run[r]);
    float rsum[8];
    for (int r = 0; r < 8; ++r)
      rsum[r] = sc[0][r] + sc[1][r] + sc[2][r] + sc[3][r];
    for (int m2 = 1; m2 <= 8; m2 <<= 1)
      for (int r = 0; r < 8; ++r)
        rsum[r] += __shfl_xor(rsum[r], m2, 32);
    for (int r = 0; r < 8; ++r) l_run[r] += rsum[r];
    for (int ns = 0; ns < 4; ++ns)
      for (int r = 0; r < 8; ++r)
        o[ns][r] *= pscale[r];

    // P to wave-private LDS, wave-local dscnt sync, then P @ V^T
    _Float16* pw = Ps + w * 16 * 64;
    for (int ns = 0; ns < 4; ++ns)
      for (int r = 0; r < 8; ++r)
        pw[(r + hi8) * 64 + ns * 16 + ln] = (_Float16)sc[ns][r];
    asm volatile("s_wait_dscnt 0" ::: "memory");

    v16h pa0 = lds_frag(pw, ln, 64, 0,  hi8);
    v16h pa1 = lds_frag(pw, ln, 64, 32, hi8);
    for (int dv = 0; dv < 4; ++dv) {
      v16h vb0 = lds_frag(Vt, dv * 16 + ln, 64, 0,  hi8);
      v16h vb1 = lds_frag(Vt, dv * 16 + ln, 64, 32, hi8);
      o[dv] = wmma_f16(pa0, vb0, o[dv]);
      o[dv] = wmma_f16(pa1, vb1, o[dv]);
    }
  }

  float inv[8];
  for (int r = 0; r < 8; ++r) inv[r] = 1.0f / l_run[r];
  _Float16* og = out + ((size_t)b * SS + qt * 64 + w * 16) * SD + h * DVV;
  for (int ns = 0; ns < 4; ++ns)
    for (int r = 0; r < 8; ++r)
      og[(size_t)(hi8 + r) * SD + ns * 16 + ln] = (_Float16)(o[ns][r] * inv[r]);
}

// ---------------------------------------------------------------------------
// LayerNorm with fused residual: x = LN(x + a)*g + b; also writes f16 mirror.
// One wave per row (512/32 = 16 elements per lane).
// ---------------------------------------------------------------------------
__global__ __launch_bounds__(256) void ln_kernel(
    float* __restrict__ x, _Float16* __restrict__ xh,
    const float* __restrict__ a, const float* __restrict__ g,
    const float* __restrict__ bt) {
  const int lane = threadIdx.x & 31;
  const int wave = threadIdx.x >> 5;
  const int row  = blockIdx.x * 8 + wave;
  float* xr = x + (size_t)row * DD;
  const float* ar = a + (size_t)row * DD;
  _Float16* hr = xh + (size_t)row * DD;

  float vals[16];
  float s = 0.0f;
  for (int j = 0; j < 16; j += 4) {
    float4 fx = *(const float4*)(xr + lane * 16 + j);
    float4 fa = *(const float4*)(ar + lane * 16 + j);
    vals[j + 0] = fx.x + fa.x; vals[j + 1] = fx.y + fa.y;
    vals[j + 2] = fx.z + fa.z; vals[j + 3] = fx.w + fa.w;
    s += vals[j] + vals[j + 1] + vals[j + 2] + vals[j + 3];
  }
  for (int m2 = 1; m2 <= 16; m2 <<= 1) s += __shfl_xor(s, m2, 32);
  float mu = s * (1.0f / DD);
  float vs = 0.0f;
  for (int j = 0; j < 16; ++j) { float d = vals[j] - mu; vs += d * d; }
  for (int m2 = 1; m2 <= 16; m2 <<= 1) vs += __shfl_xor(vs, m2, 32);
  float rs = rsqrtf(vs * (1.0f / DD) + 1e-5f);
  for (int j = 0; j < 16; ++j) {
    int c = lane * 16 + j;
    float y = (vals[j] - mu) * rs * g[c] + bt[c];
    xr[c] = y;
    hr[c] = (_Float16)y;
  }
}

// ---------------------------------------------------------------------------
// Host orchestration
// ---------------------------------------------------------------------------
extern "C" void kernel_launch(void* const* d_in, const int* in_sizes, int n_in,
                              void* d_out, int out_size, void* d_ws, size_t ws_size,
                              hipStream_t stream) {
  const int*   src   = (const int*)d_in[0];
  const int*   tgt   = (const int*)d_in[1];
  const float* emb   = (const float*)d_in[2];
  const float* e_wq  = (const float*)d_in[3];
  const float* e_wk  = (const float*)d_in[4];
  const float* e_wv  = (const float*)d_in[5];
  const float* e_wo  = (const float*)d_in[6];
  const float* e_g1  = (const float*)d_in[7];
  const float* e_b1  = (const float*)d_in[8];
  const float* e_w1  = (const float*)d_in[9];
  const float* e_bb1 = (const float*)d_in[10];
  const float* e_w2  = (const float*)d_in[11];
  const float* e_bb2 = (const float*)d_in[12];
  const float* e_g2  = (const float*)d_in[13];
  const float* e_b2  = (const float*)d_in[14];
  const float* d_swq = (const float*)d_in[15];
  const float* d_swk = (const float*)d_in[16];
  const float* d_swv = (const float*)d_in[17];
  const float* d_swo = (const float*)d_in[18];
  const float* d_g1  = (const float*)d_in[19];
  const float* d_b1  = (const float*)d_in[20];
  const float* d_cwq = (const float*)d_in[21];
  const float* d_cwk = (const float*)d_in[22];
  const float* d_cwv = (const float*)d_in[23];
  const float* d_cwo = (const float*)d_in[24];
  const float* d_g2  = (const float*)d_in[25];
  const float* d_b2  = (const float*)d_in[26];
  const float* d_w1  = (const float*)d_in[27];
  const float* d_bb1 = (const float*)d_in[28];
  const float* d_w2  = (const float*)d_in[29];
  const float* d_bb2 = (const float*)d_in[30];
  const float* d_g3  = (const float*)d_in[31];
  const float* d_b3  = (const float*)d_in[32];
  const float* out_w = (const float*)d_in[33];
  const float* out_b = (const float*)d_in[34];

  const size_t ACT = (size_t)BB * SS * DD;       // 2,097,152
  char* wsb = (char*)d_ws;
  size_t off = 0;
  auto alloc_f32 = [&](size_t n) {
    float* p = (float*)(wsb + off); off = (off + n * 4 + 255) & ~(size_t)255; return p;
  };
  auto alloc_f16 = [&](size_t n) {
    _Float16* p = (_Float16*)(wsb + off); off = (off + n * 2 + 255) & ~(size_t)255; return p;
  };

  float*    enc_x  = alloc_f32(ACT);
  float*    dec_x  = alloc_f32(ACT);
  float*    pb     = alloc_f32(ACT);             // fp32 proj/FFN2 output
  _Float16* enc_xh = alloc_f16(ACT);
  _Float16* dec_xh = alloc_f16(ACT);
  _Float16* qh     = alloc_f16(ACT);
  _Float16* kh     = alloc_f16(ACT);
  _Float16* vth    = alloc_f16(ACT);             // [B,H,64,S]
  _Float16* t0h    = alloc_f16(ACT);
  _Float16* ffnh   = alloc_f16((size_t)BB * SS * DFF);

  // f16 transposed weights (once per call)
  auto maketr = [&](const float* w, int K, int N) {
    _Float16* p = alloc_f16((size_t)K * N);
    wt_kernel<<<(int)((size_t)K * N / 256), 256, 0, stream>>>(w, p, K, N);
    return p;
  };
  const size_t WQK = (size_t)DD * (HH * DKK);
  const size_t WF1 = (size_t)DD * DFF;
  const size_t WF2 = (size_t)DFF * DD;

  _Float16 *te_wq[NLL], *te_wk[NLL], *te_wv[NLL], *te_wo[NLL], *te_w1[NLL], *te_w2[NLL];
  _Float16 *td_swq[NLL], *td_swk[NLL], *td_swv[NLL], *td_swo[NLL];
  _Float16 *td_cwq[NLL], *td_cwk[NLL], *td_cwv[NLL], *td_cwo[NLL];
  _Float16 *td_w1[NLL], *td_w2[NLL];
  for (int l = 0; l < NLL; ++l) {
    te_wq[l] = maketr(e_wq + l * WQK, DD, HH * DKK);
    te_wk[l] = maketr(e_wk + l * WQK, DD, HH * DKK);
    te_wv[l] = maketr(e_wv + l * WQK, DD, HH * DVV);
    te_wo[l] = maketr(e_wo + l * WQK, HH * DVV, DD);
    te_w1[l] = maketr(e_w1 + l * WF1, DD, DFF);
    te_w2[l] = maketr(e_w2 + l * WF2, DFF, DD);
    td_swq[l] = maketr(d_swq + l * WQK, DD, HH * DKK);
    td_swk[l] = maketr(d_swk + l * WQK, DD, HH * DKK);
    td_swv[l] = maketr(d_swv + l * WQK, DD, HH * DVV);
    td_swo[l] = maketr(d_swo + l * WQK, HH * DVV, DD);
    td_cwq[l] = maketr(d_cwq + l * WQK, DD, HH * DKK);
    td_cwk[l] = maketr(d_cwk + l * WQK, DD, HH * DKK);
    td_cwv[l] = maketr(d_cwv + l * WQK, DD, HH * DVV);
    td_cwo[l] = maketr(d_cwo + l * WQK, HH * DVV, DD);
    td_w1[l] = maketr(d_w1 + l * WF1, DD, DFF);
    td_w2[l] = maketr(d_w2 + l * WF2, DFF, DD);
  }
  _Float16* out_wt = maketr(out_w, DD, VOC);

  const int M = BB * SS;                          // 4096
  auto gemm = [&](const _Float16* A, const _Float16* Wt, const float* bias,
                  void* out, int K, int N, int relu, int mode) {
    dim3 grid(N / 64, M / 64);
    gemm_kernel<<<grid, 256, 0, stream>>>(A, Wt, bias, out, M, K, N, relu, mode);
  };
  auto ln = [&](float* x, _Float16* xh, const float* a, const float* g,
                const float* b) {
    ln_kernel<<<M / 8, 256, 0, stream>>>(x, xh, a, g, b);
  };
  dim3 agrid(SS / 64, HH, BB);

  embed_pe_kernel<<<(int)(ACT / 256), 256, 0, stream>>>(src, emb, enc_x, enc_xh);
  embed_pe_kernel<<<(int)(ACT / 256), 256, 0, stream>>>(tgt, emb, dec_x, dec_xh);

  // ---- Encoder ----
  for (int l = 0; l < NLL; ++l) {
    gemm(enc_xh, te_wq[l], nullptr, qh,  DD, HH * DKK, 0, 1);
    gemm(enc_xh, te_wk[l], nullptr, kh,  DD, HH * DKK, 0, 1);
    gemm(enc_xh, te_wv[l], nullptr, vth, DD, HH * DVV, 0, 2);
    attn_kernel<<<agrid, 128, 0, stream>>>(qh, kh, vth, t0h, 0);
    gemm(t0h, te_wo[l], nullptr, pb, HH * DVV, DD, 0, 0);
    ln(enc_x, enc_xh, pb, e_g1 + l * DD, e_b1 + l * DD);
    gemm(enc_xh, te_w1[l], e_bb1 + l * DFF, ffnh, DD, DFF, 1, 1);
    gemm(ffnh, te_w2[l], e_bb2 + l * DD, pb, DFF, DD, 0, 0);
    ln(enc_x, enc_xh, pb, e_g2 + l * DD, e_b2 + l * DD);
  }

  // ---- Decoder ----
  for (int l = 0; l < NLL; ++l) {
    gemm(dec_xh, td_swq[l], nullptr, qh,  DD, HH * DKK, 0, 1);
    gemm(dec_xh, td_swk[l], nullptr, kh,  DD, HH * DKK, 0, 1);
    gemm(dec_xh, td_swv[l], nullptr, vth, DD, HH * DVV, 0, 2);
    attn_kernel<<<agrid, 128, 0, stream>>>(qh, kh, vth, t0h, 1);
    gemm(t0h, td_swo[l], nullptr, pb, HH * DVV, DD, 0, 0);
    ln(dec_x, dec_xh, pb, d_g1 + l * DD, d_b1 + l * DD);

    gemm(dec_xh, td_cwq[l], nullptr, qh,  DD, HH * DKK, 0, 1);
    gemm(enc_xh, td_cwk[l], nullptr, kh,  DD, HH * DKK, 0, 1);
    gemm(enc_xh, td_cwv[l], nullptr, vth, DD, HH * DVV, 0, 2);
    attn_kernel<<<agrid, 128, 0, stream>>>(qh, kh, vth, t0h, 0);
    gemm(t0h, td_cwo[l], nullptr, pb, HH * DVV, DD, 0, 0);
    ln(dec_x, dec_xh, pb, d_g2 + l * DD, d_b2 + l * DD);

    gemm(dec_xh, td_w1[l], d_bb1 + l * DFF, ffnh, DD, DFF, 1, 1);
    gemm(ffnh, td_w2[l], d_bb2 + l * DD, pb, DFF, DD, 0, 0);
    ln(dec_x, dec_xh, pb, d_g3 + l * DD, d_b3 + l * DD);
  }

  // ---- Output projection: logits [B,S,V] fp32 ----
  gemm(dec_xh, out_wt, out_b, d_out, DD, VOC, 0, 0);
}
